// GTNdeepseek_40132174414158
// MI455X (gfx1250) — compile-verified, matrix operand order checked by reading, not run on hardware
//
#include <hip/hip_runtime.h>
#include <hip/hip_bf16.h>

// ---------------- CDNA5 WMMA types ----------------
typedef __attribute__((ext_vector_type(16))) __bf16 v16bf;
typedef __attribute__((ext_vector_type(8)))  float  v8f;

#define N_NODES  6144
#define IN_DIM   64
#define D_MODEL  128
#define N_HEADS  4
#define DH       32
#define FF_DIM   512
#define N_LAYERS 2
#define N_EDGES  196608
#define N_PAIRS  100000

__device__ inline v8f v8f_zero() {
  v8f z;
#pragma unroll
  for (int i = 0; i < 8; ++i) z[i] = 0.0f;
  return z;
}

__device__ inline v8f wmma_bf16(v16bf a, v16bf b, v8f c) {
  // D = A(16x32) * B(32x16) + C, f32 accumulate
  return __builtin_amdgcn_wmma_f32_16x16x32_bf16(false, a, false, b, (short)0, c,
                                                 false, false);
}

// Load a 16x32 bf16 A/B fragment from a row-major [rows][stride] bf16 buffer.
// Lane mapping per ISA 7.12.2: lanes 0-15 row=row0+lane, K=0..7 & 16..23;
// lanes 16-31 same rows, K=8..15 & 24..31.  Two 16B loads per lane.
__device__ inline v16bf frag_ld(const __hip_bfloat16* p, int row0, int stride, int col0) {
  const int lane = threadIdx.x & 31;
  const int r    = row0 + (lane & 15);
  const int kh   = (lane >> 4) << 3;
  union { uint4 u[2]; v16bf v; } f;
  const __hip_bfloat16* q = p + (size_t)r * stride + col0 + kh;
  f.u[0] = *(const uint4*)q;
  f.u[1] = *(const uint4*)(q + 16);
  return f.v;
}

// ---------------- small utility kernels ----------------
__global__ void k_cvt_bf16(const float* __restrict__ in, __hip_bfloat16* __restrict__ out, int n) {
  int i = blockIdx.x * blockDim.x + threadIdx.x;
  if (i < n) out[i] = __float2bfloat16(in[i]);
}

// out[n*K + k] = in[k*Nc + n]  (weights -> transposed bf16 [Nc][K])
__global__ void k_cvtT_bf16(const float* __restrict__ in, __hip_bfloat16* __restrict__ out,
                            int K, int Nc) {
  int i = blockIdx.x * blockDim.x + threadIdx.x;
  if (i < K * Nc) {
    int n = i / K, k = i - n * K;
    out[i] = __float2bfloat16(in[(size_t)k * Nc + n]);
  }
}

__global__ void k_zero_f4(float4* __restrict__ p, size_t n4) {
  size_t i = (size_t)blockIdx.x * blockDim.x + threadIdx.x;
  if (i < n4) p[i] = make_float4(0.f, 0.f, 0.f, 0.f);
}

__global__ void k_scatter_adj(const int* __restrict__ es, const int* __restrict__ ed,
                              const float* __restrict__ ew, float* __restrict__ adj, int n) {
  int e = blockIdx.x * blockDim.x + threadIdx.x;
  if (e < n) adj[(size_t)es[e] * N_NODES + ed[e]] = ew[e];
}

__global__ void k_copy_f32(const float* __restrict__ in, float* __restrict__ out, int n) {
  int i = blockIdx.x * blockDim.x + threadIdx.x;
  if (i < n) out[i] = in[i];
}

// ---------------- WMMA GEMM: C[M,Nc] = act(A[M,K] @ BT[Nc,K]^T + bias) -----
// 64x64 tile per 128-thread block (4 waves, each 32x32 = 2x2 WMMA frags).
template <bool BIAS, bool RELU, bool OUTF, bool OUTB>
__global__ __launch_bounds__(128) void k_gemm_bf16(
    const __hip_bfloat16* __restrict__ A, const __hip_bfloat16* __restrict__ BT,
    const float* __restrict__ bias, float* __restrict__ Cf,
    __hip_bfloat16* __restrict__ Cb, int Nc, int K) {
  __shared__ __hip_bfloat16 As[64 * 32];
  __shared__ __hip_bfloat16 Bs[64 * 32];
  const int tid = threadIdx.x, lane = tid & 31, wave = tid >> 5;
  const int m0 = blockIdx.x * 64, n0 = blockIdx.y * 64;
  const int wr = (wave >> 1) * 32, wc = (wave & 1) * 32;

  v8f acc[2][2];
#pragma unroll
  for (int i = 0; i < 2; ++i)
#pragma unroll
    for (int j = 0; j < 2; ++j) acc[i][j] = v8f_zero();

  for (int k0 = 0; k0 < K; k0 += 32) {
    __syncthreads();
#pragma unroll
    for (int i = tid; i < 256; i += 128) {  // 64 rows x 4 x uint4 per operand
      const int row = i >> 2, q = i & 3;
      *(uint4*)&As[row * 32 + q * 8] = *(const uint4*)&A[(size_t)(m0 + row) * K + k0 + q * 8];
      *(uint4*)&Bs[row * 32 + q * 8] = *(const uint4*)&BT[(size_t)(n0 + row) * K + k0 + q * 8];
    }
    if (k0 + 32 < K) {  // gfx1250 global_prefetch of next K-slab
      __builtin_prefetch(&A[(size_t)(m0 + (tid & 63)) * K + k0 + 32], 0, 1);
      __builtin_prefetch(&BT[(size_t)(n0 + (tid & 63)) * K + k0 + 32], 0, 1);
    }
    __syncthreads();

    v16bf af[2], bfr[2];
    af[0]  = frag_ld(As, wr, 32, 0);
    af[1]  = frag_ld(As, wr + 16, 32, 0);
    bfr[0] = frag_ld(Bs, wc, 32, 0);
    bfr[1] = frag_ld(Bs, wc + 16, 32, 0);
#pragma unroll
    for (int i = 0; i < 2; ++i)
#pragma unroll
      for (int j = 0; j < 2; ++j) acc[i][j] = wmma_bf16(af[i], bfr[j], acc[i][j]);
  }

  const int mh = (lane >> 4) << 3, nl = lane & 15;
#pragma unroll
  for (int i = 0; i < 2; ++i)
#pragma unroll
    for (int j = 0; j < 2; ++j) {
      const int col = n0 + wc + j * 16 + nl;
      const float bv = BIAS ? bias[col] : 0.0f;
#pragma unroll
      for (int r = 0; r < 8; ++r) {
        const int row = m0 + wr + i * 16 + mh + r;
        float v = acc[i][j][r] + bv;
        if (RELU) v = fmaxf(v, 0.0f);
        if (OUTF) Cf[(size_t)row * Nc + col] = v;
        if (OUTB) Cb[(size_t)row * Nc + col] = __float2bfloat16(v);
      }
    }
}

// ---------------- fused flash attention with dense adjacency bias ---------
// grid = (N/64 query tiles, H heads), block = 128 (4 waves, 16 q-rows each).
__global__ __launch_bounds__(128) void k_flash_attn(
    const __hip_bfloat16* __restrict__ qb, const __hip_bfloat16* __restrict__ kb,
    const __hip_bfloat16* __restrict__ vb, const float* __restrict__ adj,
    __hip_bfloat16* __restrict__ ob, float scale) {
  __shared__ __hip_bfloat16 Vt[DH * 64];        // V^T tile: [dh][key]
  __shared__ __hip_bfloat16 Pl[4][16 * 64];     // per-wave softmax P (A-layout bounce)
  const int tid = threadIdx.x, lane = tid & 31, wave = tid >> 5;
  const int qbase = blockIdx.x * 64, head = blockIdx.y, hoff = head * DH;
  const int qrow0 = qbase + wave * 16;
  const int mh = (lane >> 4) << 3, nl = lane & 15;

  // Q fragment (16 rows x DH=32) lives in registers for the whole key loop.
  const v16bf qf = frag_ld(qb, qrow0, D_MODEL, hoff);

  float mold[8], lsum[8];
#pragma unroll
  for (int r = 0; r < 8; ++r) { mold[r] = -3.0e38f; lsum[r] = 0.0f; }
  v8f acc[2] = {v8f_zero(), v8f_zero()};

  for (int kt = 0; kt < N_NODES / 64; ++kt) {
    const int kbase = kt * 64;

    // stage V^T into LDS (transpose on store)
    __syncthreads();
#pragma unroll
    for (int i = tid; i < 256; i += 128) {
      const int row = i >> 2, q = i & 3;
      union { uint4 u; __hip_bfloat16 h[8]; } d;
      d.u = *(const uint4*)&vb[(size_t)(kbase + row) * D_MODEL + hoff + q * 8];
#pragma unroll
      for (int e = 0; e < 8; ++e) Vt[(q * 8 + e) * 64 + row] = d.h[e];
    }
    __syncthreads();

    // S = Q K^T (K fragments loaded straight from global: B layout == row-major [key][dh])
    v8f s[4];
#pragma unroll
    for (int t = 0; t < 4; ++t)
      s[t] = wmma_bf16(qf, frag_ld(kb, kbase + t * 16, D_MODEL, hoff), v8f_zero());

    // scale + adjacency bias
#pragma unroll
    for (int t = 0; t < 4; ++t)
#pragma unroll
      for (int r = 0; r < 8; ++r) {
        const int qr = qrow0 + mh + r, kc = kbase + t * 16 + nl;
        s[t][r] = s[t][r] * scale + adj[(size_t)qr * N_NODES + kc];
      }

    // online softmax (row reductions across 16 lanes of each half-wave)
    float rmax[8];
#pragma unroll
    for (int r = 0; r < 8; ++r) {
      float m = s[0][r];
#pragma unroll
      for (int t = 1; t < 4; ++t) m = fmaxf(m, s[t][r]);
      m = fmaxf(m, __shfl_xor(m, 1, 16));
      m = fmaxf(m, __shfl_xor(m, 2, 16));
      m = fmaxf(m, __shfl_xor(m, 4, 16));
      m = fmaxf(m, __shfl_xor(m, 8, 16));
      rmax[r] = m;
    }
    float alpha[8], rsum[8];
#pragma unroll
    for (int r = 0; r < 8; ++r) {
      const float mn = fmaxf(mold[r], rmax[r]);
      alpha[r] = __expf(mold[r] - mn);
      mold[r] = mn;
      rsum[r] = 0.0f;
    }
#pragma unroll
    for (int t = 0; t < 4; ++t)
#pragma unroll
      for (int r = 0; r < 8; ++r) {
        const float pv = __expf(s[t][r] - mold[r]);
        s[t][r] = pv;
        rsum[r] += pv;
      }
#pragma unroll
    for (int r = 0; r < 8; ++r) {
      float sm = rsum[r];
      sm += __shfl_xor(sm, 1, 16);
      sm += __shfl_xor(sm, 2, 16);
      sm += __shfl_xor(sm, 4, 16);
      sm += __shfl_xor(sm, 8, 16);
      lsum[r] = lsum[r] * alpha[r] + sm;
    }
#pragma unroll
    for (int ns = 0; ns < 2; ++ns)
#pragma unroll
      for (int r = 0; r < 8; ++r) acc[ns][r] *= alpha[r];

    // P: C-layout -> A-layout via per-wave LDS (same-wave DS ops are in order)
    __hip_bfloat16* pw = &Pl[wave][0];
#pragma unroll
    for (int t = 0; t < 4; ++t)
#pragma unroll
      for (int r = 0; r < 8; ++r)
        pw[(mh + r) * 64 + t * 16 + nl] = __float2bfloat16(s[t][r]);

    // O += P (16x64) @ V (64x32): 2 K-steps x 2 N-subtiles
#pragma unroll
    for (int ks = 0; ks < 2; ++ks) {
      const v16bf pf = frag_ld(pw, 0, 64, ks * 32);
#pragma unroll
      for (int ns = 0; ns < 2; ++ns)
        acc[ns] = wmma_bf16(pf, frag_ld(Vt, ns * 16, 64, ks * 32), acc[ns]);
    }
  }

  // normalize and emit bf16 head slice
#pragma unroll
  for (int ns = 0; ns < 2; ++ns)
#pragma unroll
    for (int r = 0; r < 8; ++r) {
      const float o = acc[ns][r] / lsum[r];
      const int row = qrow0 + mh + r, col = hoff + ns * 16 + nl;
      ob[(size_t)row * D_MODEL + col] = __float2bfloat16(o);
    }
}

// ---------------- residual + LayerNorm (wave-per-row) ----------------------
__global__ __launch_bounds__(256) void k_res_ln(float* __restrict__ h,
                                                const float* __restrict__ y,
                                                const float* __restrict__ g,
                                                const float* __restrict__ b,
                                                __hip_bfloat16* __restrict__ hb) {
  const int row = blockIdx.x * 8 + (threadIdx.x >> 5);
  const int lane = threadIdx.x & 31;
  float v[4];
  float sum = 0.0f;
#pragma unroll
  for (int i = 0; i < 4; ++i) {
    const int c = lane + i * 32;
    v[i] = h[(size_t)row * D_MODEL + c] + y[(size_t)row * D_MODEL + c];
    sum += v[i];
  }
#pragma unroll
  for (int off = 1; off < 32; off <<= 1) sum += __shfl_xor(sum, off, 32);
  const float mu = sum * (1.0f / D_MODEL);
  float var = 0.0f;
#pragma unroll
  for (int i = 0; i < 4; ++i) { const float d = v[i] - mu; var += d * d; }
#pragma unroll
  for (int off = 1; off < 32; off <<= 1) var += __shfl_xor(var, off, 32);
  const float rs = rsqrtf(var * (1.0f / D_MODEL) + 1e-5f);
#pragma unroll
  for (int i = 0; i < 4; ++i) {
    const int c = lane + i * 32;
    const float o = (v[i] - mu) * rs * g[c] + b[c];
    h[(size_t)row * D_MODEL + c] = o;
    hb[(size_t)row * D_MODEL + c] = __float2bfloat16(o);
  }
}

// ---------------- link predictor MLP --------------------------------------
__global__ __launch_bounds__(256) void k_predictor(
    const float* __restrict__ h, const int* __restrict__ ps, const int* __restrict__ pd,
    const int* __restrict__ ns_, const int* __restrict__ nd,
    const float* __restrict__ Wp1, const float* __restrict__ bp1,
    const float* __restrict__ Wp2, const float* __restrict__ bp2,
    float* __restrict__ out) {
  __shared__ float sW[D_MODEL * 64];
  __shared__ float sb1[64], sW2[64];
  __shared__ float sb2;
  const int tid = threadIdx.x;
  for (int i = tid; i < D_MODEL * 64; i += 256) sW[i] = Wp1[i];
  if (tid < 64) { sb1[tid] = bp1[tid]; sW2[tid] = Wp2[tid]; }
  if (tid == 0) sb2 = bp2[0];
  __syncthreads();

  const int p = blockIdx.x * 256 + tid;
  if (p >= 2 * N_PAIRS) return;
  const int src = (p < N_PAIRS) ? ps[p] : ns_[p - N_PAIRS];
  const int dst = (p < N_PAIRS) ? pd[p] : nd[p - N_PAIRS];
  const float* hs = h + (size_t)src * D_MODEL;
  const float* hd = h + (size_t)dst * D_MODEL;

  float acc[64];
#pragma unroll
  for (int j = 0; j < 64; ++j) acc[j] = sb1[j];
  for (int k = 0; k < D_MODEL; ++k) {
    const float zk = hs[k] * hd[k];
    const float* wrow = &sW[k * 64];
#pragma unroll
    for (int j = 0; j < 64; ++j) acc[j] += zk * wrow[j];
  }
  float score = sb2;
#pragma unroll
  for (int j = 0; j < 64; ++j) {
    float z1 = acc[j];
    z1 = (z1 > 0.0f) ? z1 : 0.2f * z1;  // leaky relu, slope 0.2
    score += z1 * sW2[j];
  }
  out[p] = score;
}

// ---------------- host orchestration --------------------------------------
extern "C" void kernel_launch(void* const* d_in, const int* in_sizes, int n_in,
                              void* d_out, int out_size, void* d_ws, size_t ws_size,
                              hipStream_t stream) {
  (void)in_sizes; (void)n_in; (void)out_size; (void)ws_size;
  const float* x        = (const float*)d_in[0];
  const int* edge_src   = (const int*)d_in[1];
  const int* edge_dst   = (const int*)d_in[2];
  const float* edge_w   = (const float*)d_in[3];
  const int* pos_src    = (const int*)d_in[4];
  const int* pos_dst    = (const int*)d_in[5];
  const int* neg_src    = (const int*)d_in[6];
  const int* neg_dst    = (const int*)d_in[7];
  const float* W_in     = (const float*)d_in[8];
  const float* b_in     = (const float*)d_in[9];
  const float* Wq       = (const float*)d_in[10];
  const float* Wk       = (const float*)d_in[11];
  const float* Wv       = (const float*)d_in[12];
  const float* Wo       = (const float*)d_in[13];
  const float* ln1_g    = (const float*)d_in[14];
  const float* ln1_b    = (const float*)d_in[15];
  const float* ln2_g    = (const float*)d_in[16];
  const float* ln2_b    = (const float*)d_in[17];
  const float* W1       = (const float*)d_in[18];
  const float* b1       = (const float*)d_in[19];
  const float* W2       = (const float*)d_in[20];
  const float* b2       = (const float*)d_in[21];
  const float* Wp1      = (const float*)d_in[22];
  const float* bp1      = (const float*)d_in[23];
  const float* Wp2      = (const float*)d_in[24];
  const float* bp2      = (const float*)d_in[25];
  float* out            = (float*)d_out;

  // workspace layout (all offsets 256B aligned)
  unsigned char* w = (unsigned char*)d_ws;
  float*          h  = (float*)w;           w += (size_t)N_NODES * D_MODEL * 4;
  __hip_bfloat16* hb = (__hip_bfloat16*)w;  w += (size_t)N_NODES * D_MODEL * 2;
  __hip_bfloat16* xb = (__hip_bfloat16*)w;  w += (size_t)N_NODES * IN_DIM * 2;
  __hip_bfloat16* qb = (__hip_bfloat16*)w;  w += (size_t)N_NODES * D_MODEL * 2;
  __hip_bfloat16* kb = (__hip_bfloat16*)w;  w += (size_t)N_NODES * D_MODEL * 2;
  __hip_bfloat16* vb = (__hip_bfloat16*)w;  w += (size_t)N_NODES * D_MODEL * 2;
  __hip_bfloat16* ob = (__hip_bfloat16*)w;  w += (size_t)N_NODES * D_MODEL * 2;
  __hip_bfloat16* fb = (__hip_bfloat16*)w;  w += (size_t)N_NODES * FF_DIM * 2;
  float*          y  = (float*)w;           w += (size_t)N_NODES * D_MODEL * 4;
  __hip_bfloat16* wT = (__hip_bfloat16*)w;  w += (size_t)FF_DIM * D_MODEL * 2;
  float*          adj = (float*)w;          // N*N*4 = 151 MB

  const float scale = 0.17677669529663687f;  // 1/sqrt(DH)
  const dim3 gGemmD(N_NODES / 64, D_MODEL / 64);   // (96,2)
  const dim3 gGemmF(N_NODES / 64, FF_DIM / 64);    // (96,8)
  const dim3 gAttn(N_NODES / 64, N_HEADS);         // (96,4)

  // input projection: h = x @ W_in + b_in  (f32 + bf16 mirror)
  k_cvt_bf16<<<(N_NODES * IN_DIM + 255) / 256, 256, 0, stream>>>(x, xb, N_NODES * IN_DIM);
  k_cvtT_bf16<<<(IN_DIM * D_MODEL + 255) / 256, 256, 0, stream>>>(W_in, wT, IN_DIM, D_MODEL);
  k_gemm_bf16<true, false, true, true><<<gGemmD, 128, 0, stream>>>(xb, wT, b_in, h, hb,
                                                                   D_MODEL, IN_DIM);

  for (int l = 0; l < N_LAYERS; ++l) {
    // dense adjacency bias
    const size_t n4 = (size_t)N_NODES * N_NODES / 4;
    k_zero_f4<<<(unsigned)((n4 + 255) / 256), 256, 0, stream>>>((float4*)adj, n4);
    k_scatter_adj<<<(N_EDGES + 255) / 256, 256, 0, stream>>>(
        edge_src + (size_t)l * N_EDGES, edge_dst + (size_t)l * N_EDGES,
        edge_w + (size_t)l * N_EDGES, adj, N_EDGES);

    // QKV projections (bf16 outputs for the WMMA attention)
    k_cvtT_bf16<<<(D_MODEL * D_MODEL + 255) / 256, 256, 0, stream>>>(
        Wq + (size_t)l * D_MODEL * D_MODEL, wT, D_MODEL, D_MODEL);
    k_gemm_bf16<false, false, false, true><<<gGemmD, 128, 0, stream>>>(hb, wT, nullptr,
                                                                       nullptr, qb, D_MODEL, D_MODEL);
    k_cvtT_bf16<<<(D_MODEL * D_MODEL + 255) / 256, 256, 0, stream>>>(
        Wk + (size_t)l * D_MODEL * D_MODEL, wT, D_MODEL, D_MODEL);
    k_gemm_bf16<false, false, false, true><<<gGemmD, 128, 0, stream>>>(hb, wT, nullptr,
                                                                       nullptr, kb, D_MODEL, D_MODEL);
    k_cvtT_bf16<<<(D_MODEL * D_MODEL + 255) / 256, 256, 0, stream>>>(
        Wv + (size_t)l * D_MODEL * D_MODEL, wT, D_MODEL, D_MODEL);
    k_gemm_bf16<false, false, false, true><<<gGemmD, 128, 0, stream>>>(hb, wT, nullptr,
                                                                       nullptr, vb, D_MODEL, D_MODEL);

    // fused attention with adjacency bias
    k_flash_attn<<<gAttn, 128, 0, stream>>>(qb, kb, vb, adj, ob, scale);

    // output projection + residual LN
    k_cvtT_bf16<<<(D_MODEL * D_MODEL + 255) / 256, 256, 0, stream>>>(
        Wo + (size_t)l * D_MODEL * D_MODEL, wT, D_MODEL, D_MODEL);
    k_gemm_bf16<false, false, true, false><<<gGemmD, 128, 0, stream>>>(ob, wT, nullptr, y,
                                                                       nullptr, D_MODEL, D_MODEL);
    k_res_ln<<<N_NODES / 8, 256, 0, stream>>>(h, y, ln1_g + l * D_MODEL, ln1_b + l * D_MODEL, hb);

    // FFN
    k_cvtT_bf16<<<(D_MODEL * FF_DIM + 255) / 256, 256, 0, stream>>>(
        W1 + (size_t)l * D_MODEL * FF_DIM, wT, D_MODEL, FF_DIM);
    k_gemm_bf16<true, true, false, true><<<gGemmF, 128, 0, stream>>>(
        hb, wT, b1 + (size_t)l * FF_DIM, nullptr, fb, FF_DIM, D_MODEL);
    k_cvtT_bf16<<<(FF_DIM * D_MODEL + 255) / 256, 256, 0, stream>>>(
        W2 + (size_t)l * FF_DIM * D_MODEL, wT, FF_DIM, D_MODEL);
    k_gemm_bf16<true, false, true, false><<<gGemmD, 128, 0, stream>>>(
        fb, wT, b2 + (size_t)l * D_MODEL, y, nullptr, D_MODEL, FF_DIM);
    k_res_ln<<<N_NODES / 8, 256, 0, stream>>>(h, y, ln2_g + l * D_MODEL, ln2_b + l * D_MODEL, hb);
  }

  // predictor: out[0:P)=pos, out[P:2P)=neg, then h
  k_predictor<<<(2 * N_PAIRS + 255) / 256, 256, 0, stream>>>(
      h, pos_src, pos_dst, neg_src, neg_dst, Wp1, bp1, Wp2, bp2, out);
  k_copy_f32<<<(N_NODES * D_MODEL + 255) / 256, 256, 0, stream>>>(
      h, out + 2 * N_PAIRS, N_NODES * D_MODEL);
}
